// HianModel_39702677684510
// MI455X (gfx1250) — matrix operand order repaired and checked by reading, not compiled, not run on hardware
//
#include <hip/hip_runtime.h>

// ============================================================================
// HianModel on MI455X (gfx1250): hierarchical attention, bf16 WMMA path.
// Heavy stages (word conv implicit-GEMM, word attention) use
// v_wmma_f32_16x16x32_bf16 with 2x4 accumulator tiles per wave (A reused 4x,
// B reused 2x -> 0.75 tile-loads per WMMA instead of 2).  Tiny
// sentence/aspect/result attentions are f32 scalar kernels (<<1% of FLOPs).
// Per-batch-item loop keeps the ~110MB working set inside the 192MB L2.
// Requires ws_size >= ~125 MB.
// ============================================================================

typedef __attribute__((ext_vector_type(16))) __bf16   bf16x16;
typedef __attribute__((ext_vector_type(8)))  float    f32x8;
typedef __attribute__((ext_vector_type(8)))  unsigned u32x8;

#define DEVFN __device__ __forceinline__

// ---- model dims ----
#define D_B   16
#define D_N   50
#define D_S   10
#define D_MW  25
#define D_LW  250      // words per seq (S*MW)
#define D_LWP 256      // padded to 16 tiles
#define D_WD  768
#define D_E1  512
#define D_E2  256
#define D_A   6

// ---------------------------------------------------------------------------
// bf16 helpers (RNE pack)
// ---------------------------------------------------------------------------
DEVFN unsigned short bfbits(float f) {
  unsigned u = __builtin_bit_cast(unsigned, f);
  u += 0x7fffu + ((u >> 16) & 1u);          // round-to-nearest-even
  return (unsigned short)(u >> 16);
}
DEVFN __bf16 f2bf(float f) { unsigned short h = bfbits(f); return __builtin_bit_cast(__bf16, h); }
DEVFN float  bf2f(__bf16 h) {
  unsigned short s = __builtin_bit_cast(unsigned short, h);
  unsigned u = ((unsigned)s) << 16;
  return __builtin_bit_cast(float, u);
}
DEVFN unsigned pack2f(float lo, float hi) {
  return (unsigned)bfbits(lo) | ((unsigned)bfbits(hi) << 16);
}

// ---------------------------------------------------------------------------
// WMMA fragment loaders (wave32; this code's contiguous k order, same for A and B)
//   A 16x32 bf16: lane L holds row m=L&15; K base (L>>4)*16;
//     VGPR v -> K = kb+2v (pairs packed)
//   B 32x16 bf16 (K x N): lane L holds col n=L&15, K base (L>>4)*16;
//     VGPR v -> K = kb+2v (pairs packed)
//   C/D 16x16 f32: VGPR r -> row (L>>4)*8+r, col L&15
// ---------------------------------------------------------------------------
DEVFN bf16x16 load_a_frag(const __bf16* arow, int kk, int lane) {
  u32x8 u; int kb = (lane >> 4) * 16;
#pragma unroll
  for (int v = 0; v < 8; ++v) u[v] = *(const unsigned*)(arow + kk + kb + 2 * v);
  return __builtin_bit_cast(bf16x16, u);
}
// "BT" mode: weight stored N x K row-major (torch (out,in)); contiguous pairs.
DEVFN bf16x16 load_bt_frag(const __bf16* wrow, int kk, int lane) {
  u32x8 u; int kb = (lane >> 4) * 16;
#pragma unroll
  for (int v = 0; v < 8; ++v) u[v] = *(const unsigned*)(wrow + kk + kb + 2 * v);
  return __builtin_bit_cast(bf16x16, u);
}
// "BN" mode: B stored K x N row-major (e.g. V matrix); strided pair gather.
DEVFN bf16x16 load_bn_frag(const __bf16* B, int ldb, int kk, int nc, int lane) {
  u32x8 u; int kb = (lane >> 4) * 16;
#pragma unroll
  for (int v = 0; v < 8; ++v) {
    int k = kk + kb + 2 * v;
    unsigned lo = __builtin_bit_cast(unsigned short, B[(size_t)k * ldb + nc]);
    unsigned hi = __builtin_bit_cast(unsigned short, B[(size_t)(k + 1) * ldb + nc]);
    u[v] = lo | (hi << 16);
  }
  return __builtin_bit_cast(bf16x16, u);
}
DEVFN f32x8 wmma_acc(bf16x16 a, bf16x16 b, f32x8 c) {
  return __builtin_amdgcn_wmma_f32_16x16x32_bf16(false, a, false, b, (short)0, c, false, false);
}

// One K-step for an NM x NNt accumulator grid, weight-transposed B.
template<int NM, int NNt>
DEVFN void step_bt(f32x8 (&acc)[NM][NNt], const __bf16* const (&arow)[NM],
                   const __bf16* const (&wrow)[NNt], int kk, int lane) {
  bf16x16 a[NM];
#pragma unroll
  for (int i = 0; i < NM; ++i) a[i] = load_a_frag(arow[i], kk, lane);
#pragma unroll
  for (int j = 0; j < NNt; ++j) {
    bf16x16 b = load_bt_frag(wrow[j], kk, lane);
#pragma unroll
    for (int i = 0; i < NM; ++i) acc[i][j] = wmma_acc(a[i], b, acc[i][j]);
  }
}
// One K-step, row-major KxN B.
template<int NM, int NNt>
DEVFN void step_bn(f32x8 (&acc)[NM][NNt], const __bf16* const (&arow)[NM],
                   const __bf16* B, int ldb, const int (&nc)[NNt], int kk, int lane) {
  bf16x16 a[NM];
#pragma unroll
  for (int i = 0; i < NM; ++i) a[i] = load_a_frag(arow[i], kk, lane);
#pragma unroll
  for (int j = 0; j < NNt; ++j) {
    bf16x16 b = load_bn_frag(B, ldb, kk, nc[j], lane);
#pragma unroll
    for (int i = 0; i < NM; ++i) acc[i][j] = wmma_acc(a[i], b, acc[i][j]);
  }
}

// ---------------------------------------------------------------------------
// Weight conversion / packing kernels (run once per launch)
// ---------------------------------------------------------------------------
__global__ void k_f32_to_bf16(const float* __restrict__ src, __bf16* __restrict__ dst, int n) {
  int i = blockIdx.x * 256 + threadIdx.x;
  if (i < n) dst[i] = f2bf(src[i]);
}
// conv weight (O, C, KT) row-major -> packed [o][kt][c] bf16 so the GEMM
// K-index = kt*C + c is contiguous in c for BT fragment loads.
__global__ void k_pack_convw(const float* __restrict__ w, __bf16* __restrict__ dst,
                             int O, int C, int KT) {
  int i = blockIdx.x * 256 + threadIdx.x;
  int tot = O * C * KT;
  if (i >= tot) return;
  int o = i / (C * KT); int rem = i % (C * KT);
  int kt = rem / C;     int c = rem % C;
  dst[i] = f2bf(w[((size_t)o * C + c) * KT + kt]);
}

// ---------------------------------------------------------------------------
// Conv1d (k=3, same-pad) as implicit GEMM.  A rows = positions l (f32 input,
// staged to LDS as bf16 pairs, shared by all 4 waves), B = packed weights.
// Block 128 = 4 waves x 4 ntiles each = 16 ntiles (256 out channels) per
// block.  grid.x = nSeq*MTILES, grid.y = O/256.  All index math templated
// so it constant-folds (no SALU magic-division sequences).
// Stores ALL tile rows; rows >= L stored as 0 (zero padding downstream).
// ---------------------------------------------------------------------------
template<int MTILES, int CC, int OO, int LL>
__global__ void __launch_bounds__(128)
k_conv_wmma(const float* __restrict__ x, int xStrideN,
            const __bf16* __restrict__ wB, const float* __restrict__ bias,
            __bf16* outB, int outBStrideN, float* outF, int outFStrideN) {
  constexpr int Ktot = 3 * CC;
  int tid = threadIdx.x, wave = tid >> 5, lane = tid & 31;
  int n = blockIdx.x / MTILES, mtile = blockIdx.x % MTILES;   // MTILES is constexpr
  const float* xn = x + (size_t)n * xStrideN;

  __shared__ unsigned lsA[256];     // 16 rows x 16 bf16-pairs
  f32x8 acc[4] = {};
  const __bf16* wrow[4];
#pragma unroll
  for (int j = 0; j < 4; ++j) {
    int ntile = blockIdx.y * 16 + wave * 4 + j;
    wrow[j] = wB + (size_t)(ntile * 16 + (lane & 15)) * Ktot;
  }

  for (int kk = 0; kk < Ktot; kk += 32) {
    int kt = kk / CC, c0 = kk % CC;   // CC % 32 == 0 -> chunk within one tap
    for (int u = tid; u < 256; u += 128) {
      int row = u >> 4, pr = u & 15;
      int l = mtile * 16 + row;
      int ls = l + kt - 1;            // pad = 1 for k=3
      float f0 = 0.f, f1 = 0.f;
      if (ls >= 0 && ls < LL) {
        const float* p = xn + (size_t)ls * CC + (c0 + pr * 2);
        f0 = p[0]; f1 = p[1];
        __builtin_prefetch(p + CC, 0, 0);   // next tap / next chunk line
      }
      lsA[u] = pack2f(f0, f1);
    }
    __syncthreads();
    u32x8 au; int m = lane & 15, kb8 = (lane >> 4) * 8;
#pragma unroll
    for (int v = 0; v < 8; ++v) au[v] = lsA[m * 16 + kb8 + v];
    bf16x16 a = __builtin_bit_cast(bf16x16, au);
#pragma unroll
    for (int j = 0; j < 4; ++j) {
      bf16x16 b = load_bt_frag(wrow[j], kk, lane);
      acc[j] = wmma_acc(a, b, acc[j]);
    }
    __syncthreads();
  }
  int col = lane & 15;
#pragma unroll
  for (int j = 0; j < 4; ++j) {
    int oc = (blockIdx.y * 16 + wave * 4 + j) * 16 + col;
    float bv = bias[oc];
#pragma unroll
    for (int r = 0; r < 8; ++r) {
      int l = mtile * 16 + ((lane >> 4) * 8) + r;
      float val = (l < LL) ? fmaxf(acc[j][r] + bv, 0.f) : 0.f;
      if (outB) outB[(size_t)n * outBStrideN + (size_t)l * OO + oc] = f2bf(val);
      if (outF) outF[(size_t)n * outFStrideN + (size_t)l * OO + oc] = val;
    }
  }
}

// ---------------------------------------------------------------------------
// QKV GEMM (flattened 12800 x 512 x 1536): qkv = h * W^T + b, bf16 out.
// Wave: 2 mtiles x 4 ntiles.  Block: 8 mtiles x 4 ntiles.
// grid.x = 12800/16/8 = 100, grid.y = 1536/64 = 24.
// ---------------------------------------------------------------------------
__global__ void __launch_bounds__(128)
k_gemm_qkv(const __bf16* __restrict__ A, const __bf16* __restrict__ W,
           const float* __restrict__ bias, __bf16* __restrict__ out) {
  int tid = threadIdx.x, wave = tid >> 5, lane = tid & 31;
  int mbase = (blockIdx.x * 4 + wave) * 2;
  f32x8 acc[2][4] = {};
  const __bf16* arow[2];
  const __bf16* wrow[4];
#pragma unroll
  for (int i = 0; i < 2; ++i)
    arow[i] = A + (size_t)((mbase + i) * 16 + (lane & 15)) * D_E1;
#pragma unroll
  for (int j = 0; j < 4; ++j)
    wrow[j] = W + (size_t)((blockIdx.y * 4 + j) * 16 + (lane & 15)) * D_E1;
#pragma unroll 2
  for (int kk = 0; kk < D_E1; kk += 32) {
    __builtin_prefetch(arow[0] + kk + 128, 0, 0);
    step_bt<2, 4>(acc, arow, wrow, kk, lane);
  }
#pragma unroll
  for (int i = 0; i < 2; ++i)
#pragma unroll
    for (int j = 0; j < 4; ++j) {
      int oc = (blockIdx.y * 4 + j) * 16 + (lane & 15);
      float bv = bias[oc];
#pragma unroll
      for (int r = 0; r < 8; ++r) {
        int row = (mbase + i) * 16 + ((lane >> 4) * 8) + r;
        out[(size_t)row * (3 * D_E1) + oc] = f2bf(acc[i][j][r] + bv);
      }
    }
}

// ---------------------------------------------------------------------------
// Scores: s[n,l,l'] = q[n,l,:] . k[n,l',:] / sqrt(512).  Per-seq B matrix.
// Wave: 2 mtiles x 4 ntiles; block covers 8 mtiles.  grid.x = 50*2, y = 4.
// ---------------------------------------------------------------------------
__global__ void __launch_bounds__(128)
k_gemm_qk(const __bf16* __restrict__ qkv, float* __restrict__ s) {
  int tid = threadIdx.x, wave = tid >> 5, lane = tid & 31;
  int n = blockIdx.x >> 1, mbase = (blockIdx.x & 1) * 8 + wave * 2;
  const __bf16* base = qkv + (size_t)n * D_LWP * (3 * D_E1);
  f32x8 acc[2][4] = {};
  const __bf16* arow[2];
  const __bf16* wrow[4];
#pragma unroll
  for (int i = 0; i < 2; ++i)
    arow[i] = base + (size_t)((mbase + i) * 16 + (lane & 15)) * (3 * D_E1);           // q
#pragma unroll
  for (int j = 0; j < 4; ++j)
    wrow[j] = base + (size_t)((blockIdx.y * 4 + j) * 16 + (lane & 15)) * (3 * D_E1) + D_E1; // k
#pragma unroll 2
  for (int kk = 0; kk < D_E1; kk += 32)
    step_bt<2, 4>(acc, arow, wrow, kk, lane);
  const float scale = 0.04419417382415922f; // 1/sqrt(512)
#pragma unroll
  for (int i = 0; i < 2; ++i)
#pragma unroll
    for (int j = 0; j < 4; ++j) {
      int col = (blockIdx.y * 4 + j) * 16 + (lane & 15);
#pragma unroll
      for (int r = 0; r < 8; ++r) {
        int row = (mbase + i) * 16 + ((lane >> 4) * 8) + r;
        s[((size_t)n * D_LWP + row) * D_LWP + col] = acc[i][j][r] * scale;
      }
    }
}

// ---------------------------------------------------------------------------
// Row softmax over 250 valid cols; pad cols/rows -> 0; bf16 output P.
// grid = 50*256 blocks, block 256.
// ---------------------------------------------------------------------------
__global__ void __launch_bounds__(256)
k_softmax(const float* __restrict__ s, __bf16* __restrict__ p) {
  int n = blockIdx.x >> 8, l = blockIdx.x & 255;
  int j = threadIdx.x;
  __shared__ float red[256];
  bool valid = (l < D_LW) && (j < D_LW);
  float sv = valid ? s[((size_t)n * D_LWP + l) * D_LWP + j] : -1e30f;
  red[j] = sv; __syncthreads();
  for (int st = 128; st > 0; st >>= 1) { if (j < st) red[j] = fmaxf(red[j], red[j + st]); __syncthreads(); }
  float mx = red[0]; __syncthreads();
  float e = valid ? __expf(sv - mx) : 0.f;
  red[j] = e; __syncthreads();
  for (int st = 128; st > 0; st >>= 1) { if (j < st) red[j] += red[j + st]; __syncthreads(); }
  float sum = red[0];
  float pv = (l < D_LW && sum > 0.f) ? e / sum : 0.f;
  p[((size_t)n * D_LWP + l) * D_LWP + j] = f2bf(pv);
}

// ---------------------------------------------------------------------------
// O = P * V  (per-seq, BN-mode B).  Wave: 2 mtiles x 4 ntiles.
// grid.x = 50*2, grid.y = 512/64 = 8.
// ---------------------------------------------------------------------------
__global__ void __launch_bounds__(128)
k_gemm_pv(const __bf16* __restrict__ p, const __bf16* __restrict__ qkv,
          __bf16* __restrict__ o) {
  int tid = threadIdx.x, wave = tid >> 5, lane = tid & 31;
  int n = blockIdx.x >> 1, mbase = (blockIdx.x & 1) * 8 + wave * 2;
  const __bf16* vb = qkv + (size_t)n * D_LWP * (3 * D_E1) + 2 * D_E1;
  f32x8 acc[2][4] = {};
  const __bf16* arow[2];
  int nc[4];
#pragma unroll
  for (int i = 0; i < 2; ++i)
    arow[i] = p + ((size_t)n * D_LWP + (mbase + i) * 16 + (lane & 15)) * D_LWP;
#pragma unroll
  for (int j = 0; j < 4; ++j)
    nc[j] = (blockIdx.y * 4 + j) * 16 + (lane & 15);
#pragma unroll 2
  for (int kk = 0; kk < D_LWP; kk += 32)
    step_bn<2, 4>(acc, arow, vb, 3 * D_E1, nc, kk, lane);
#pragma unroll
  for (int i = 0; i < 2; ++i)
#pragma unroll
    for (int j = 0; j < 4; ++j)
#pragma unroll
      for (int r = 0; r < 8; ++r) {
        int row = (mbase + i) * 16 + ((lane >> 4) * 8) + r;
        o[((size_t)n * D_LWP + row) * D_E1 + nc[j]] = f2bf(acc[i][j][r]);
      }
}

// ---------------------------------------------------------------------------
// Out-proj fused with elementwise gate: h2 = h * (o*W^T + b).  Flattened M.
// grid.x = 100, grid.y = 8.
// ---------------------------------------------------------------------------
__global__ void __launch_bounds__(128)
k_outproj(const __bf16* __restrict__ o, const __bf16* __restrict__ W,
          const float* __restrict__ bias, const __bf16* __restrict__ h,
          float* __restrict__ h2) {
  int tid = threadIdx.x, wave = tid >> 5, lane = tid & 31;
  int mbase = (blockIdx.x * 4 + wave) * 2;
  f32x8 acc[2][4] = {};
  const __bf16* arow[2];
  const __bf16* wrow[4];
#pragma unroll
  for (int i = 0; i < 2; ++i)
    arow[i] = o + (size_t)((mbase + i) * 16 + (lane & 15)) * D_E1;
#pragma unroll
  for (int j = 0; j < 4; ++j)
    wrow[j] = W + (size_t)((blockIdx.y * 4 + j) * 16 + (lane & 15)) * D_E1;
#pragma unroll 2
  for (int kk = 0; kk < D_E1; kk += 32)
    step_bt<2, 4>(acc, arow, wrow, kk, lane);
#pragma unroll
  for (int i = 0; i < 2; ++i)
#pragma unroll
    for (int j = 0; j < 4; ++j) {
      int oc = (blockIdx.y * 4 + j) * 16 + (lane & 15);
      float bv = bias[oc];
#pragma unroll
      for (int r = 0; r < 8; ++r) {
        int row = (mbase + i) * 16 + ((lane >> 4) * 8) + r;
        size_t idx = (size_t)row * D_E1 + oc;
        h2[idx] = bf2f(h[idx]) * (acc[i][j][r] + bv);
      }
    }
}

// ---------------------------------------------------------------------------
// sent[n,s,e] = sum_{w<25} h2[n, s*25+w, e]
// ---------------------------------------------------------------------------
__global__ void k_sentsum(const float* __restrict__ h2, float* __restrict__ sent) {
  int n = blockIdx.x / D_S, s = blockIdx.x % D_S;
  for (int e = threadIdx.x; e < D_E1; e += 256) {
    float acc = 0.f;
#pragma unroll
    for (int w = 0; w < D_MW; ++w)
      acc += h2[((size_t)n * D_LWP + s * D_MW + w) * D_E1 + e];
    sent[((size_t)n * D_S + s) * D_E1 + e] = acc;
  }
}

// ---------------------------------------------------------------------------
// Small-attention helpers (f32 scalar; <1% of FLOPs)
// ---------------------------------------------------------------------------
__global__ void k_small_qkv(const float* __restrict__ x, int sb, int sl,
                            const float* __restrict__ w, const float* __restrict__ b,
                            float* __restrict__ out, int L, int E) {
  int bid = blockIdx.x;
  int bi = bid / L, l = bid % L;
  const float* xr = x + (size_t)bi * sb + (size_t)l * sl;
  for (int f = threadIdx.x; f < 3 * E; f += 256) {
    const float* wr = w + (size_t)f * E;
    float acc = b[f];
    for (int e = 0; e < E; ++e) acc += xr[e] * wr[e];
    out[(size_t)bid * 3 * E + f] = acc;
  }
}
__global__ void k_small_attn(const float* __restrict__ qkv, float* __restrict__ o,
                             int L, int E) {
  int bid = blockIdx.x;
  int bi = bid / L;
  int E3 = 3 * E;
  const float* q = qkv + (size_t)bid * E3;
  __shared__ float sp[64];
  int j = threadIdx.x;
  float sv = -1e30f;
  if (j < L) {
    const float* kr = qkv + ((size_t)bi * L + j) * E3 + E;
    float acc = 0.f;
    for (int e = 0; e < E; ++e) acc += q[e] * kr[e];
    sv = acc * rsqrtf((float)E);
  }
  sp[j] = sv; __syncthreads();
  float mx = -1e30f;
  for (int t = 0; t < L; ++t) mx = fmaxf(mx, sp[t]);
  __syncthreads();
  float pv = (j < L) ? __expf(sv - mx) : 0.f;
  sp[j] = pv; __syncthreads();
  float sum = 0.f;
  for (int t = 0; t < L; ++t) sum += sp[t];
  float inv = 1.f / sum;
  for (int e = j; e < E; e += 64) {
    float acc = 0.f;
    for (int t = 0; t < L; ++t)
      acc += sp[t] * qkv[((size_t)bi * L + t) * E3 + 2 * E + e];
    o[(size_t)bid * E + e] = acc * inv;
  }
}
__global__ void k_small_proj(const float* __restrict__ o, const float* __restrict__ w,
                             const float* __restrict__ b, float* __restrict__ out, int E) {
  int bid = blockIdx.x, f = threadIdx.x;
  const float* orow = o + (size_t)bid * E;
  const float* wr = w + (size_t)f * E;
  float acc = b[f];
  for (int e = 0; e < E; ++e) acc += orow[e] * wr[e];
  out[(size_t)bid * E + f] = acc;
}

// sc[n,s,:] *= saOut[s,n,:]        (sc padded 16 rows per n)
__global__ void k_scmul(float* __restrict__ sc, const float* __restrict__ saOut) {
  int n = blockIdx.x / D_S, s = blockIdx.x % D_S, e = threadIdx.x;
  sc[((size_t)n * 16 + s) * D_E2 + e] *= saOut[((size_t)s * D_N + n) * D_E2 + e];
}
// asp[n,a,:] = sum_{s: g[n,s]==a} sc[n,s,:]
__global__ void k_asp(const float* __restrict__ sc, const long long* __restrict__ groups,
                      float* __restrict__ asp) {
  int n = blockIdx.x, e = threadIdx.x;
  float acc[D_A] = {0.f, 0.f, 0.f, 0.f, 0.f, 0.f};
  for (int s = 0; s < D_S; ++s) {
    int g = (int)groups[(size_t)n * D_S + s];
    if (g >= 0 && g < D_A) acc[g] += sc[((size_t)n * 16 + s) * D_E2 + e];
  }
#pragma unroll
  for (int a = 0; a < D_A; ++a) asp[((size_t)n * D_A + a) * D_E2 + e] = acc[a];
}
// r[n,:] = sum_a asp[n,a,:] * aaOut[a,n,:]
__global__ void k_rred(const float* __restrict__ asp, const float* __restrict__ aaOut,
                       float* __restrict__ r) {
  int n = blockIdx.x, e = threadIdx.x;
  float acc = 0.f;
#pragma unroll
  for (int a = 0; a < D_A; ++a)
    acc += asp[((size_t)n * D_A + a) * D_E2 + e] * aaOut[((size_t)a * D_N + n) * D_E2 + e];
  r[(size_t)n * D_E2 + e] = acc;
}
// out[b,n,:] = r[n,:] * raOut[n,:]
__global__ void k_final(const float* __restrict__ r, const float* __restrict__ raOut,
                        float* __restrict__ out) {
  int n = blockIdx.x, e = threadIdx.x;
  out[(size_t)n * D_E2 + e] = r[(size_t)n * D_E2 + e] * raOut[(size_t)n * D_E2 + e];
}

// ===========================================================================
// Host launcher
// ===========================================================================
static inline size_t alignUp(size_t x) { return (x + 255) & ~(size_t)255; }

extern "C" void kernel_launch(void* const* d_in, const int* in_sizes, int n_in,
                              void* d_out, int out_size, void* d_ws, size_t ws_size,
                              hipStream_t stream) {
  (void)in_sizes; (void)n_in; (void)out_size; (void)ws_size;
  const float*     xs       = (const float*)d_in[0];
  const long long* groups   = (const long long*)d_in[1];
  const float*     wconv_w  = (const float*)d_in[2];
  const float*     wconv_b  = (const float*)d_in[3];
  const float*     sconv_w  = (const float*)d_in[4];
  const float*     sconv_b  = (const float*)d_in[5];
  const float*     wa_in_w  = (const float*)d_in[6];
  const float*     wa_in_b  = (const float*)d_in[7];
  const float*     wa_out_w = (const float*)d_in[8];
  const float*     wa_out_b = (const float*)d_in[9];
  const float*     sa_in_w  = (const float*)d_in[10];
  const float*     sa_in_b  = (const float*)d_in[11];
  const float*     sa_out_w = (const float*)d_in[12];
  const float*     sa_out_b = (const float*)d_in[13];
  const float*     aa_in_w  = (const float*)d_in[14];
  const float*     aa_in_b  = (const float*)d_in[15];
  const float*     aa_out_w = (const float*)d_in[16];
  const float*     aa_out_b = (const float*)d_in[17];
  const float*     ra_in_w  = (const float*)d_in[18];
  const float*     ra_in_b  = (const float*)d_in[19];
  const float*     ra_out_w = (const float*)d_in[20];
  const float*     ra_out_b = (const float*)d_in[21];
  float* dout = (float*)d_out;

  // ---- workspace carve-up ----
  char* ws = (char*)d_ws;
  size_t off = 0;
  auto carve = [&](size_t bytes) { char* p = ws + off; off = alignUp(off + bytes); return p; };
  __bf16* wconvB  = (__bf16*)carve((size_t)D_E1 * 3 * D_WD * 2);
  __bf16* sconvB  = (__bf16*)carve((size_t)D_E2 * 3 * D_E1 * 2);
  __bf16* waInWb  = (__bf16*)carve((size_t)3 * D_E1 * D_E1 * 2);
  __bf16* waOutWb = (__bf16*)carve((size_t)D_E1 * D_E1 * 2);
  __bf16* hB      = (__bf16*)carve((size_t)D_N * D_LWP * D_E1 * 2);
  __bf16* qkvB    = (__bf16*)carve((size_t)D_N * D_LWP * 3 * D_E1 * 2);
  float*  sbuf    = (float*) carve((size_t)D_N * D_LWP * D_LWP * 4);
  __bf16* pB      = (__bf16*)carve((size_t)D_N * D_LWP * D_LWP * 2);
  __bf16* oB      = (__bf16*)carve((size_t)D_N * D_LWP * D_E1 * 2);
  float*  h2      = (float*) carve((size_t)D_N * D_LWP * D_E1 * 4);
  float*  sent    = (float*) carve((size_t)D_N * D_S * D_E1 * 4);
  float*  sc      = (float*) carve((size_t)D_N * 16 * D_E2 * 4);
  float*  sqkv    = (float*) carve((size_t)D_S * D_N * 3 * D_E2 * 4);
  float*  so      = (float*) carve((size_t)D_S * D_N * D_E2 * 4);
  float*  sout    = (float*) carve((size_t)D_S * D_N * D_E2 * 4);
  float*  asp     = (float*) carve((size_t)D_N * D_A * D_E2 * 4);
  float*  rbuf    = (float*) carve((size_t)D_N * D_E2 * 4);
  float*  raOut   = (float*) carve((size_t)D_N * D_E2 * 4);

  // ---- one-time weight conversion (bf16 / conv repack) ----
  k_f32_to_bf16<<<(3 * D_E1 * D_E1 + 255) / 256, 256, 0, stream>>>(wa_in_w, waInWb, 3 * D_E1 * D_E1);
  k_f32_to_bf16<<<(D_E1 * D_E1 + 255) / 256, 256, 0, stream>>>(wa_out_w, waOutWb, D_E1 * D_E1);
  k_pack_convw<<<(D_E1 * D_WD * 3 + 255) / 256, 256, 0, stream>>>(wconv_w, wconvB, D_E1, D_WD, 3);
  k_pack_convw<<<(D_E2 * D_E1 * 3 + 255) / 256, 256, 0, stream>>>(sconv_w, sconvB, D_E2, D_E1, 3);

  // ---- per-item pipeline (working set stays L2-resident) ----
  for (int b = 0; b < D_B; ++b) {
    const float* xb = xs + (size_t)b * D_N * D_LW * D_WD;
    const long long* gb = groups + (size_t)b * D_N * D_S;

    // word conv 768->512 (implicit GEMM, K=2304), relu -> h (bf16, padded rows)
    k_conv_wmma<16, D_WD, D_E1, D_LW><<<dim3(D_N * 16, D_E1 / 256), 128, 0, stream>>>(
        xb, D_LW * D_WD, wconvB, wconv_b, hB, D_LWP * D_E1, nullptr, 0);
    // word attention
    k_gemm_qkv<<<dim3(100, 24), 128, 0, stream>>>(hB, waInWb, wa_in_b, qkvB);
    k_gemm_qk<<<dim3(D_N * 2, 4), 128, 0, stream>>>(qkvB, sbuf);
    k_softmax<<<D_N * D_LWP, 256, 0, stream>>>(sbuf, pB);
    k_gemm_pv<<<dim3(D_N * 2, 8), 128, 0, stream>>>(pB, qkvB, oB);
    k_outproj<<<dim3(100, 8), 128, 0, stream>>>(oB, waOutWb, wa_out_b, hB, h2);
    // sentence sum + conv 512->256
    k_sentsum<<<D_N * D_S, 256, 0, stream>>>(h2, sent);
    k_conv_wmma<1, D_E1, D_E2, D_S><<<dim3(D_N, 1), 128, 0, stream>>>(
        sent, D_S * D_E1, sconvB, sconv_b, nullptr, 0, sc, 16 * D_E2);
    // sentence attention over N (batch = s): x(bi=s, l=n) = sc[(n*16+s)*E2]
    k_small_qkv<<<D_S * D_N, 256, 0, stream>>>(sc, D_E2, 16 * D_E2, sa_in_w, sa_in_b, sqkv, D_N, D_E2);
    k_small_attn<<<D_S * D_N, 64, 0, stream>>>(sqkv, so, D_N, D_E2);
    k_small_proj<<<D_S * D_N, D_E2, 0, stream>>>(so, sa_out_w, sa_out_b, sout, D_E2);
    k_scmul<<<D_N * D_S, D_E2, 0, stream>>>(sc, sout);
    // aspect gather + aspect attention over N (batch = a)
    k_asp<<<D_N, D_E2, 0, stream>>>(sc, gb, asp);
    k_small_qkv<<<D_A * D_N, 256, 0, stream>>>(asp, D_E2, D_A * D_E2, aa_in_w, aa_in_b, sqkv, D_N, D_E2);
    k_small_attn<<<D_A * D_N, 64, 0, stream>>>(sqkv, so, D_N, D_E2);
    k_small_proj<<<D_A * D_N, D_E2, 0, stream>>>(so, aa_out_w, aa_out_b, sout, D_E2);
    k_rred<<<D_N, D_E2, 0, stream>>>(asp, sout, rbuf);
    // result attention over N (batch = 1)
    k_small_qkv<<<D_N, 256, 0, stream>>>(rbuf, 0, D_E2, ra_in_w, ra_in_b, sqkv, D_N, D_E2);
    k_small_attn<<<D_N, 64, 0, stream>>>(sqkv, so, D_N, D_E2);
    k_small_proj<<<D_N, D_E2, 0, stream>>>(so, ra_out_w, ra_out_b, raOut, D_E2);
    k_final<<<D_N, D_E2, 0, stream>>>(rbuf, raOut, dout + (size_t)b * D_N * D_E2);
  }
}
